// Gtu2d_40724879900789
// MI455X (gfx1250) — matrix operand order repaired and checked by reading, not compiled, not run on hardware
//
#include <hip/hip_runtime.h>
#include <cstdint>
#include <cstddef>

// ---------------------------------------------------------------------------
// Gtu2d for MI455X (gfx1250, wave32).
//   GEMMs (u/v/out projections + RPE MLP)  -> bf16 WMMA 16x16x32, f32 accum
//   2-D depthwise Toeplitz operator        -> 128x128 radix-2 FFT conv (VALU)
//   v-channel fetch in fft_conv            -> TDM tensor_load_to_lds (async)
// FFT size 128 >= 111 lags => no circular aliasing vs the reference's 112.
// ---------------------------------------------------------------------------

#define B_    8
#define EMB   512
#define D1    1024
#define NH    8
#define HD    128
#define RPE   512
#define NGRID 111
#define NPOS  (111*111)     // 12321
#define SPAT  (56*56)       // 3136
#define TOK   (B_*SPAT)     // 25088
#define LFFT  128
#define PITCH 129           // odd pitch: conflict-free column FFTs
#define LDSBYTES_G    (2*128*PITCH*4)              // 132096 bytes (SR + SI)
#define LDSBYTES_CONV (2*128*PITCH*4 + SPAT*4)     // + 12544B TDM staging

typedef __attribute__((ext_vector_type(16))) __bf16 v16bf;
typedef __attribute__((ext_vector_type(8)))  float  v8f;
typedef unsigned int __attribute__((ext_vector_type(4))) u32x4;
typedef int          __attribute__((ext_vector_type(4))) i32x4;
typedef int          __attribute__((ext_vector_type(8))) i32x8;

#if defined(__has_builtin)
#if __has_builtin(__builtin_amdgcn_tensor_load_to_lds) && \
    __has_builtin(__builtin_amdgcn_s_wait_tensorcnt)
#define HAVE_TDM 1
#endif
#endif

__device__ __forceinline__ unsigned short f2bf(float f) {
  unsigned int u = __float_as_uint(f);
  u += 0x7fffu + ((u >> 16) & 1u);          // round-to-nearest-even
  return (unsigned short)(u >> 16);
}
__device__ __forceinline__ float bf2f(unsigned short h) {
  return __uint_as_float(((unsigned int)h) << 16);
}

#ifdef HAVE_TDM
// Issue one TDM copy of `nelem` contiguous f32 from global `gaddr` into LDS
// byte-offset `lds_off`. D# per CDNA5 ISA ch.8: one 2-D tile of shape
// (tile_dim1=1 x tile_dim0=nelem), data_size=4B, type=2 ("image").
// Wave-level op (EXEC ignored); tracked by TENSORcnt.
__device__ __forceinline__ void tdm_load_f32(unsigned lds_off, const void* gaddr,
                                             int nelem) {
  unsigned long long ga = (unsigned long long)(uintptr_t)gaddr;
  u32x4 g0;
  g0[0] = 1u;                                        // count=1, user descriptor
  g0[1] = lds_off;                                   // lds_addr [63:32]
  g0[2] = (unsigned)(ga & 0xffffffffu);              // global_addr [95:64]
  g0[3] = (unsigned)((ga >> 32) & 0x01ffffffu)       // global_addr [120:96]
        | (2u << 30);                                // type=2 [127:126]
  i32x8 g1;
  g1[0] = 0x20000;                                   // data_size=2 (4B), mask=0
  g1[1] = (nelem & 0xffff) << 16;                    // tensor_dim0[15:0] @63:48
  g1[2] = ((nelem >> 16) & 0xffff) | (1 << 16);      // tdim0[31:16]; tensor_dim1=1
  g1[3] = (nelem & 0xffff) << 16;                    // tile_dim0 @127:112
  g1[4] = 1;                                         // tile_dim1=1, tile_dim2=0
  g1[5] = nelem;                                     // tensor_dim0_stride[31:0]
  g1[6] = 0;
  g1[7] = 0;
  i32x4 z4 = {0, 0, 0, 0};
#if __clang_major__ >= 23
  i32x8 z8 = {0, 0, 0, 0, 0, 0, 0, 0};
  __builtin_amdgcn_tensor_load_to_lds(g0, g1, z4, z4, z8, 0);
#else
  __builtin_amdgcn_tensor_load_to_lds(g0, g1, z4, z4, 0);
#endif
}
#endif

// --------------------------- small elementwise kernels ---------------------

__global__ void convert_bf16(const float* __restrict__ in,
                             unsigned short* __restrict__ out, int n) {
  int i = blockIdx.x * 256 + threadIdx.x;
  if (i < n) out[i] = f2bf(in[i]);
}

// SimpleRMSNorm over d=512 (+ optional ReLU), emit bf16.
template<int RELU>
__global__ __launch_bounds__(256)
void normact_bf16(const float* __restrict__ in, unsigned short* __restrict__ out) {
  int row = blockIdx.x, tid = threadIdx.x;
  const float* p = in + (size_t)row * 512;
  float v0 = p[tid], v1 = p[tid + 256];
  __shared__ float red[256];
  red[tid] = v0 * v0 + v1 * v1;
  __syncthreads();
  for (int off = 128; off > 0; off >>= 1) {
    if (tid < off) red[tid] += red[tid + off];
    __syncthreads();
  }
  // x / (||x|| * d^-1/2 + eps),  1/sqrt(512) = 0.044194173824...
  float scale = 1.f / (sqrtf(red[0]) * 0.04419417382415922f + 1e-8f);
  float a0 = v0 * scale, a1 = v1 * scale;
  if (RELU) { a0 = fmaxf(a0, 0.f); a1 = fmaxf(a1, 0.f); }
  unsigned short* q = out + (size_t)row * 512;
  q[tid] = f2bf(a0);
  q[tid + 256] = f2bf(a1);
}

// t0[pos,r] = dp*W[r,0] + dq*W[r,1] + b[r]   (coords are static: dp,dq in [-55,55])
__global__ void rpe_input(const float* __restrict__ w, const float* __restrict__ b,
                          float* __restrict__ t0) {
  int idx = blockIdx.x * 256 + threadIdx.x;
  if (idx >= NPOS * RPE) return;
  int pos = idx >> 9, r = idx & 511;
  int i = pos / NGRID, jj = pos - i * NGRID;
  float dp = (float)(i - 55), dq = (float)(jj - 55);
  t0[idx] = dp * w[r * 2 + 0] + dq * w[r * 2 + 1] + b[r];
}

// --------------------------- WMMA GEMM -------------------------------------
// C[m,n] = sum_k A[m,k] * W[n,k]  (i.e. A @ W^T), A/W bf16, accum f32.
// One wave32 per 16x16 tile. EPI: 0=bias, 1=silu(bias+), 2=bias+residual.
// OUTBF: write bf16 instead of f32.
template<int EPI, int OUTBF>
__global__ __launch_bounds__(32)
void gemm16(const unsigned short* __restrict__ A,
            const unsigned short* __restrict__ Wt,
            const float* __restrict__ bias,
            const float* __restrict__ resid,
            float* __restrict__ Cf,
            unsigned short* __restrict__ Cb,
            int Mrows, int K, int N) {
  int lane = threadIdx.x;
  int l15 = lane & 15, hi = lane >> 4;
  int mt = blockIdx.x << 4, nt = blockIdx.y << 4;
  int arow = mt + l15;
  if (arow >= Mrows) arow = Mrows - 1;           // clamp; store is masked
  const unsigned short* Ap = A + (size_t)arow * K;
  const unsigned short* Wp = Wt + (size_t)(nt + l15) * K;

  v8f acc = (v8f){0.f, 0.f, 0.f, 0.f, 0.f, 0.f, 0.f, 0.f};
  for (int k = 0; k < K; k += 32) {
    // A fragment: 16-bit A 16x32 layout. lane hi selects K-octet:
    //   elems 0..7  -> K = k + hi*8 + 0..7
    //   elems 8..15 -> K = k + 16 + hi*8 + 0..7
    union { v16bf v; uint4 q[2]; } a, b;
    a.q[0] = *(const uint4*)(Ap + k + hi * 8);
    a.q[1] = *(const uint4*)(Ap + k + 16 + hi * 8);
    // B fragment: lane column n = l15, same K-octet layout as A
    // (elems 0..7 -> K = k + hi*8 + e, elems 8..15 -> K = k + 16 + hi*8 + e-8).
    b.q[0] = *(const uint4*)(Wp + k + hi * 8);
    b.q[1] = *(const uint4*)(Wp + k + 16 + hi * 8);
    __builtin_prefetch(Ap + k + 128, 0, 1);      // global_prefetch_b8
    __builtin_prefetch(Wp + k + 128, 0, 1);
    acc = __builtin_amdgcn_wmma_f32_16x16x32_bf16(
        false, a.v, false, b.v, (short)0, acc, false, false);
  }

  float bv = bias[nt + l15];
#pragma unroll
  for (int r = 0; r < 8; ++r) {
    int m = mt + hi * 8 + r;                     // C layout: VGPR r -> M = r + hi*8
    if (m >= Mrows) break;
    float xv = acc[r] + bv;
    if (EPI == 1) xv = xv / (1.f + __expf(-xv)); // silu
    size_t o = (size_t)m * N + nt + l15;
    if (EPI == 2) xv += resid[o];
    if (OUTBF) Cb[o] = f2bf(xv);
    else       Cf[o] = xv;
  }
}

// --------------------------- transpose (coalesced via LDS) -----------------
// in [rows][cols] -> out [cols][rows], batched on blockIdx.z.
__global__ __launch_bounds__(256)
void transpose_f32(const float* __restrict__ in, float* __restrict__ out,
                   int rows, int cols) {
  __shared__ float tile[32][33];
  size_t base = (size_t)blockIdx.z * rows * cols;
  int c0 = blockIdx.x * 32, r0 = blockIdx.y * 32;
  int tx = threadIdx.x, ty = threadIdx.y;
  for (int i = 0; i < 32; i += 8) {
    int r = r0 + ty + i, c = c0 + tx;
    if (r < rows && c < cols) tile[ty + i][tx] = in[base + (size_t)r * cols + c];
  }
  __syncthreads();
  for (int i = 0; i < 32; i += 8) {
    int c = c0 + ty + i, r = r0 + tx;
    if (r < rows && c < cols) out[base + (size_t)c * rows + r] = tile[tx][ty + i];
  }
}

// --------------------------- 128-pt radix-2 FFT passes ---------------------
// Forward = DIF (natural in -> bit-rev out); Inverse = DIT (bit-rev in ->
// natural out). Both spectra are in identical bit-rev order, so the pointwise
// complex product is layout-consistent and no explicit bit-reversal is needed.
__device__ void fft128_pass(float* SR, float* SI, int colPass, int inverse, int tid) {
  for (int s = 0; s < 7; ++s) {
    int logHalf = inverse ? s : (6 - s);
    int half = 1 << logHalf;
    float angScale = (inverse ? 3.14159265358979f : -3.14159265358979f) / (float)half;
    __syncthreads();
    for (int idx = tid; idx < 128 * 64; idx += 256) {
      int line = idx >> 6, j64 = idx & 63;
      int j = j64 & (half - 1);
      int bb = j64 >> logHalf;
      int p0 = (bb << (logHalf + 1)) + j;
      int p1 = p0 + half;
      int i0 = colPass ? p0 * PITCH + line : line * PITCH + p0;
      int i1 = colPass ? p1 * PITCH + line : line * PITCH + p1;
      float ur = SR[i0], ui = SI[i0], vr = SR[i1], vi = SI[i1];
      float sw, cw;
      __sincosf(angScale * (float)j, &sw, &cw);
      if (!inverse) {                            // DIF: (u+v), (u-v)*w
        float dr = ur - vr, di = ui - vi;
        SR[i0] = ur + vr;            SI[i0] = ui + vi;
        SR[i1] = dr * cw - di * sw;  SI[i1] = dr * sw + di * cw;
      } else {                                   // DIT: u + v*w, u - v*w
        float tr = vr * cw - vi * sw, ti = vr * sw + vi * cw;
        SR[i0] = ur + tr;  SI[i0] = ui + ti;
        SR[i1] = ur - tr;  SI[i1] = ui - ti;
      }
    }
  }
  __syncthreads();
}

// Forward 2-D FFT of one g channel (111x111 lags scattered into 128x128).
__global__ __launch_bounds__(256)
void fft_g(const float* __restrict__ g2, float2* __restrict__ Gf) {
  extern __shared__ float smem[];
  float* SR = smem;
  float* SI = smem + 128 * PITCH;
  int chan = blockIdx.x, tid = threadIdx.x;
  for (int i = tid; i < 128 * PITCH; i += 256) { SR[i] = 0.f; SI[i] = 0.f; }
  __syncthreads();
  const float* gp = g2 + (size_t)chan * NPOS;
  for (int idx = tid; idx < NPOS; idx += 256) {
    int i = idx / NGRID, jj = idx - i * NGRID;
    int r = (i - 55) & 127, c = (jj - 55) & 127;   // circulant placement
    SR[r * PITCH + c] = gp[idx];
  }
  fft128_pass(SR, SI, 0, 0, tid);                  // rows
  fft128_pass(SR, SI, 1, 0, tid);                  // cols
  float2* out = Gf + (size_t)chan * 16384;
  for (int idx = tid; idx < 16384; idx += 256) {
    int r = idx >> 7, c = idx & 127;
    out[idx] = make_float2(SR[r * PITCH + c], SI[r * PITCH + c]);
  }
}

// FFT(v) * Gf -> iFFT -> y, one (b, h*128+d) channel per workgroup.
// The 12.5KB v channel is DMA'd into LDS by the TDM while the block zeroes
// the FFT scratch; completion via s_wait_tensorcnt + barrier.
__global__ __launch_bounds__(256)
void fft_conv(const float* __restrict__ v2, const float2* __restrict__ Gf,
              float* __restrict__ y2) {
  extern __shared__ float smem[];
  float* SR = smem;
  float* SI = smem + 128 * PITCH;
  int gc = blockIdx.x, tid = threadIdx.x;
  int j = gc & (D1 - 1);
  const float* vp = v2 + (size_t)gc * SPAT;

#ifdef HAVE_TDM
  float* stage = smem + 2 * 128 * PITCH;           // SPAT f32 staging
  if (tid < 32) {                                  // wave 0 issues the DMA
    tdm_load_f32((unsigned)(uintptr_t)stage, vp, SPAT);
  }
#endif
  for (int i = tid; i < 128 * PITCH; i += 256) { SR[i] = 0.f; SI[i] = 0.f; }
  __syncthreads();
#ifdef HAVE_TDM
  if (tid < 32) __builtin_amdgcn_s_wait_tensorcnt(0);
  __syncthreads();
  for (int idx = tid; idx < SPAT; idx += 256) {
    int p = idx / 56, q = idx - p * 56;
    SR[p * PITCH + q] = stage[idx];
  }
#else
  for (int idx = tid; idx < SPAT; idx += 256) {
    int p = idx / 56, q = idx - p * 56;
    SR[p * PITCH + q] = vp[idx];
  }
#endif
  fft128_pass(SR, SI, 0, 0, tid);
  fft128_pass(SR, SI, 1, 0, tid);
  const float2* gf = Gf + (size_t)j * 16384;
  for (int idx = tid; idx < 16384; idx += 256) {
    int r = idx >> 7, c = idx & 127;
    int o = r * PITCH + c;
    float2 w = gf[idx];
    float ar = SR[o], ai = SI[o];
    SR[o] = ar * w.x - ai * w.y;
    SI[o] = ar * w.y + ai * w.x;
  }
  fft128_pass(SR, SI, 1, 1, tid);                  // inverse cols
  fft128_pass(SR, SI, 0, 1, tid);                  // inverse rows
  float* yp = y2 + (size_t)gc * SPAT;
  const float sc = 1.0f / 16384.f;
  for (int idx = tid; idx < SPAT; idx += 256) {
    int p = idx / 56, q = idx - p * 56;
    yp[idx] = SR[p * PITCH + q] * sc;
  }
}

// uy[t,j] = bf16( u[t,j] * y2[b,j,s] ), fused transpose via LDS.
__global__ __launch_bounds__(256)
void mul_uy(const unsigned short* __restrict__ u, const float* __restrict__ y2,
            unsigned short* __restrict__ uy) {
  __shared__ float tile[32][33];
  int b = blockIdx.z;
  int jt = blockIdx.y * 32;
  int st = blockIdx.x * 32;
  int tx = threadIdx.x, ty = threadIdx.y;
  for (int i = 0; i < 32; i += 8) {
    int j = jt + ty + i, s = st + tx;
    tile[ty + i][tx] = y2[((size_t)(b * D1 + j)) * SPAT + s];
  }
  __syncthreads();
  for (int i = 0; i < 32; i += 8) {
    int s = st + ty + i;
    size_t o = ((size_t)(b * SPAT + s)) * D1 + jt + tx;
    uy[o] = f2bf(bf2f(u[o]) * tile[tx][ty + i]);
  }
}

// --------------------------- host launcher ---------------------------------

extern "C" void kernel_launch(void* const* d_in, const int* in_sizes, int n_in,
                              void* d_out, int out_size, void* d_ws, size_t ws_size,
                              hipStream_t stream) {
  (void)in_sizes; (void)n_in; (void)out_size; (void)ws_size;
  const float* x        = (const float*)d_in[0];
  const float* W_u      = (const float*)d_in[1];
  const float* b_u      = (const float*)d_in[2];
  const float* W_v      = (const float*)d_in[3];
  const float* b_v      = (const float*)d_in[4];
  const float* W_o      = (const float*)d_in[5];
  const float* b_o      = (const float*)d_in[6];
  const float* rpe_in_w = (const float*)d_in[7];
  const float* rpe_in_b = (const float*)d_in[8];
  const float* rpe_h_w  = (const float*)d_in[9];
  const float* rpe_h_b  = (const float*)d_in[10];
  const float* rpe_out_w= (const float*)d_in[11];
  const float* rpe_out_b= (const float*)d_in[12];
  float* out = (float*)d_out;

  char* ws = (char*)d_ws;
  size_t off = 0;
  auto take = [&](size_t bytes) { size_t r = off; off += (bytes + 255) & ~(size_t)255; return r; };
  unsigned short* wWu = (unsigned short*)(ws + take((size_t)D1 * EMB * 2));
  unsigned short* wWv = (unsigned short*)(ws + take((size_t)D1 * EMB * 2));
  unsigned short* wWo = (unsigned short*)(ws + take((size_t)EMB * D1 * 2));
  unsigned short* wRh = (unsigned short*)(ws + take((size_t)3 * RPE * RPE * 2));
  unsigned short* wRo = (unsigned short*)(ws + take((size_t)D1 * RPE * 2));
  unsigned short* xn  = (unsigned short*)(ws + take((size_t)TOK * EMB * 2));
  unsigned short* u_b = (unsigned short*)(ws + take((size_t)TOK * D1 * 2));
  float* v_f          = (float*)(ws + take((size_t)TOK * D1 * 4));  // later reused as y2
  float* v2           = (float*)(ws + take((size_t)TOK * D1 * 4));  // later reused as uy
  float* t0           = (float*)(ws + take((size_t)NPOS * RPE * 4));
  unsigned short* tb  = (unsigned short*)(ws + take((size_t)NPOS * RPE * 2));
  float* g            = (float*)(ws + take((size_t)NPOS * D1 * 4));
  float* g2           = (float*)(ws + take((size_t)NPOS * D1 * 4));
  float2* Gf          = (float2*)(ws + take((size_t)D1 * 16384 * 8));
  float* y2           = v_f;                       // v_f dead after transpose
  unsigned short* uy  = (unsigned short*)v2;       // v2 dead after fft_conv

  // 0) weights -> bf16
  convert_bf16<<<(D1 * EMB + 255) / 256, 256, 0, stream>>>(W_u, wWu, D1 * EMB);
  convert_bf16<<<(D1 * EMB + 255) / 256, 256, 0, stream>>>(W_v, wWv, D1 * EMB);
  convert_bf16<<<(EMB * D1 + 255) / 256, 256, 0, stream>>>(W_o, wWo, EMB * D1);
  convert_bf16<<<(3 * RPE * RPE + 255) / 256, 256, 0, stream>>>(rpe_h_w, wRh, 3 * RPE * RPE);
  convert_bf16<<<(D1 * RPE + 255) / 256, 256, 0, stream>>>(rpe_out_w, wRo, D1 * RPE);

  // 1) xn = SimpleRMSNorm(x) (bf16)
  normact_bf16<0><<<TOK, 256, 0, stream>>>(x, xn);

  // 2) u = silu(xn @ Wu^T + bu) (bf16), v = silu(xn @ Wv^T + bv) (f32)
  dim3 guv(TOK / 16, D1 / 16);
  gemm16<1, 1><<<guv, 32, 0, stream>>>(xn, wWu, b_u, nullptr, nullptr, u_b, TOK, EMB, D1);
  gemm16<1, 0><<<guv, 32, 0, stream>>>(xn, wWv, b_v, nullptr, v_f, nullptr, TOK, EMB, D1);

  // 3) v -> channel-major v2[b*1024+j][s]
  dim3 tblk(32, 8);
  transpose_f32<<<dim3(D1 / 32, SPAT / 32, B_), tblk, 0, stream>>>(v_f, v2, SPAT, D1);

  // 4) RPE MLP -> g[pos][h*128+d]
  rpe_input<<<(NPOS * RPE + 255) / 256, 256, 0, stream>>>(rpe_in_w, rpe_in_b, t0);
  int mt_rpe = (NPOS + 15) / 16;   // 771
  for (int i = 0; i < 3; ++i) {
    normact_bf16<1><<<NPOS, 256, 0, stream>>>(t0, tb);
    gemm16<0, 0><<<dim3(mt_rpe, RPE / 16), 32, 0, stream>>>(
        tb, wRh + (size_t)i * RPE * RPE, rpe_h_b + i * RPE, nullptr, t0, nullptr,
        NPOS, RPE, RPE);
  }
  normact_bf16<1><<<NPOS, 256, 0, stream>>>(t0, tb);
  gemm16<0, 0><<<dim3(mt_rpe, D1 / 16), 32, 0, stream>>>(
      tb, wRo, rpe_out_b, nullptr, g, nullptr, NPOS, RPE, D1);

  // 5) g -> channel-major, then per-channel forward 2-D FFT -> Gf
  transpose_f32<<<dim3(D1 / 32, (NPOS + 31) / 32, 1), tblk, 0, stream>>>(g, g2, NPOS, D1);
  hipFuncSetAttribute((const void*)fft_g, hipFuncAttributeMaxDynamicSharedMemorySize, LDSBYTES_G);
  hipFuncSetAttribute((const void*)fft_conv, hipFuncAttributeMaxDynamicSharedMemorySize, LDSBYTES_CONV);
  fft_g<<<D1, 256, LDSBYTES_G, stream>>>(g2, Gf);

  // 6) y = iFFT( FFT(v) * Gf ) per (b, h, d) channel  (TDM-fed)
  fft_conv<<<B_ * D1, 256, LDSBYTES_CONV, stream>>>(v2, Gf, y2);

  // 7) uy = bf16(u * y)  (fused transpose back to token-major)
  mul_uy<<<dim3(SPAT / 32, D1 / 32, B_), tblk, 0, stream>>>(u_b, y2, uy);

  // 8) out = uy @ Wo^T + bo + x
  gemm16<2, 0><<<dim3(TOK / 16, EMB / 16), 32, 0, stream>>>(
      uy, wWo, b_o, x, out, nullptr, TOK, D1, EMB);
}
